// MultiHeadSelfAttention_19258633355622
// MI455X (gfx1250) — matrix operand order repaired and checked by reading, not compiled, not run on hardware
//
#include <hip/hip_runtime.h>
#include <hip/hip_bf16.h>
#include <math.h>

// ---------------------------------------------------------------------------
// MHSA forward for MI455X (gfx1250, wave32, WMMA 16x16x32 f16 -> f32 acc)
//   B=32, N=1024, C=768, H=12, Dh=64
// Pipeline:
//   [1] qkv_gemm : x(f32) @ w_qkv(f32) + b -> q,k (f16,(BH,N,64)) and
//                  v (f16,(BH,64,N) transposed); q pre-scaled by Dh^-0.5
//   [2] attn     : flash attention per (b,h); WMMA QK^T and PV; online softmax
//   [3] proj_gemm: attn_out(f16,(B*N,768)) @ w_proj + b_proj -> f32 out
// LDS double-buffered staging overlaps global loads with WMMA; all fragment
// reads are ds_load_b128 pairs (layouts padded to 16B alignment).
// ---------------------------------------------------------------------------

typedef __attribute__((ext_vector_type(16))) _Float16 v16h;
typedef __attribute__((ext_vector_type(8)))  _Float16 h8;
typedef __attribute__((ext_vector_type(8)))  float    v8f;

#define DMODEL 768
#define QKV_N  2304
#define SEQ_N  1024
#define NHEADS 12
#define DHEAD  64
#define BATCH  32
#define ROWS_M (BATCH * SEQ_N)                              // 32768
#define HEADELEMS ((size_t)BATCH * NHEADS * SEQ_N * DHEAD)  // 25165824

__device__ __forceinline__ v8f wmma_f16(v16h a, v16h b, v8f c) {
  return __builtin_amdgcn_wmma_f32_16x16x32_f16(false, a, false, b,
                                                (short)0, c, false, false);
}

__device__ __forceinline__ v16h cat(h8 lo, h8 hi) {
  return __builtin_shufflevector(lo, hi, 0, 1, 2, 3, 4, 5, 6, 7,
                                 8, 9, 10, 11, 12, 13, 14, 15);
}

// A-fragment 16x32 f16 (ISA 7.12.2): lane = M (lane&15), half = lane>>4.
// Register order == two contiguous 8-half chunks at K = h*8 and K = 16+h*8.
__device__ __forceinline__ v16h load_a_frag(const _Float16* src, int row0,
                                            int ld, int k0, int lane) {
  const int m = lane & 15, h = lane >> 4;
  const _Float16* p = src + (size_t)(row0 + m) * ld + k0 + h * 8;
  return cat(*(const h8*)p, *(const h8*)(p + 16));
}

// B-fragment 32x16 where B[k][n] = src[n0+n][k] (src holds B^T row-major).
// lane = N column; K = two 8-half chunks at K = h*8 and K = 16+h*8.
__device__ __forceinline__ v16h load_bT_frag(const _Float16* src, int n0,
                                             int ld, int k0, int lane) {
  const int n = lane & 15, h = lane >> 4;
  const _Float16* p = src + (size_t)(n0 + n) * ld + k0 + h * 8;
  return cat(*(const h8*)p, *(const h8*)(p + 16));
}

__device__ __forceinline__ h8 cvt8(float4 a, float4 b) {
  h8 r;
  r[0] = (_Float16)a.x; r[1] = (_Float16)a.y;
  r[2] = (_Float16)a.z; r[3] = (_Float16)a.w;
  r[4] = (_Float16)b.x; r[5] = (_Float16)b.y;
  r[6] = (_Float16)b.z; r[7] = (_Float16)b.w;
  return r;
}

// ---------------------------------------------------------------------------
// Kernel 1: QKV projection. Block = 256 threads (8 waves). Tile 256(M) x 64(N).
// Wave w owns rows [w*32, w*32+32) x 64 cols => 8 WMMA per K-step.
// Double-buffered LDS: stage K-step kk+1 while computing kk.
// ---------------------------------------------------------------------------
__global__ __launch_bounds__(256) void qkv_gemm_kernel(
    const float* __restrict__ x, const float* __restrict__ w,
    const float* __restrict__ bias,
    _Float16* __restrict__ qbuf, _Float16* __restrict__ kbuf,
    _Float16* __restrict__ vbuf) {
  __shared__ _Float16 Xs[2][256][40];   // A tiles (+8 pad, 16B-aligned rows)
  __shared__ _Float16 Wt[2][64][40];    // W tiles TRANSPOSED: Wt[n][k]

  const int t = threadIdx.x;
  const int lane = t & 31, wave = t >> 5;
  const int m0 = blockIdx.y * 256;
  const int n0 = blockIdx.x * 64;

  v8f acc[2][4];
  {
    v8f z = {};
#pragma unroll
    for (int s = 0; s < 2; ++s)
#pragma unroll
      for (int i = 0; i < 4; ++i) acc[s][i] = z;
  }

  auto stage = [&](int k0, int buf) {
#pragma unroll
    for (int c = 0; c < 4; ++c) {
      const int lin = c * 256 + t;
      const int row = lin >> 2, col = (lin & 3) * 8;
      const float* gp = x + (size_t)(m0 + row) * DMODEL + k0 + col;
      *(h8*)&Xs[buf][row][col] =
          cvt8(*(const float4*)gp, *(const float4*)(gp + 4));
    }
    const int n = t & 63, kq = (t >> 6) * 8;
    h8 hv;
#pragma unroll
    for (int i = 0; i < 8; ++i)
      hv[i] = (_Float16)w[(size_t)(k0 + kq + i) * QKV_N + n0 + n];
    *(h8*)&Wt[buf][n][kq] = hv;
    if (k0 + 64 < DMODEL)
      __builtin_prefetch(x + (size_t)(m0 + t) * DMODEL + k0 + 64, 0, 1);
  };

  stage(0, 0);
  for (int kk = 0; kk < DMODEL / 32; ++kk) {
    const int cur = kk & 1;
    __syncthreads();                       // staging of `cur` complete
    if (kk + 1 < DMODEL / 32) stage((kk + 1) * 32, cur ^ 1);

    const v16h a0 = load_a_frag(&Xs[cur][0][0], wave * 32, 40, 0, lane);
    const v16h a1 = load_a_frag(&Xs[cur][0][0], wave * 32 + 16, 40, 0, lane);
    v16h bfr[4];
#pragma unroll
    for (int tn = 0; tn < 4; ++tn)
      bfr[tn] = load_bT_frag(&Wt[cur][0][0], tn * 16, 40, 0, lane);
#pragma unroll
    for (int tn = 0; tn < 4; ++tn) {
      acc[0][tn] = wmma_f16(a0, bfr[tn], acc[0][tn]);
      acc[1][tn] = wmma_f16(a1, bfr[tn], acc[1][tn]);
    }
  }

  // Epilogue: bias, fold q-scale, scatter. q,k -> (BH,N,64); v -> (BH,64,N).
#pragma unroll
  for (int tn = 0; tn < 4; ++tn) {
    const int gcol = n0 + tn * 16 + (lane & 15);
    const int which = gcol / DMODEL;            // 0=q, 1=k, 2=v
    const int hd = gcol - which * DMODEL;
    const int head = hd >> 6, d = hd & 63;
    const float bv = bias[gcol];
    const float scale = (which == 0) ? 0.125f : 1.0f;
#pragma unroll
    for (int s = 0; s < 2; ++s) {
#pragma unroll
      for (int r = 0; r < 8; ++r) {
        const int gm = m0 + wave * 32 + s * 16 + r + 8 * (lane >> 4);
        const int bidx = gm >> 10, n = gm & 1023;
        const size_t bh = (size_t)(bidx * NHEADS + head);
        const _Float16 val = (_Float16)((acc[s][tn][r] + bv) * scale);
        if (which == 0)
          qbuf[(bh * SEQ_N + n) * DHEAD + d] = val;
        else if (which == 1)
          kbuf[(bh * SEQ_N + n) * DHEAD + d] = val;
        else
          vbuf[(bh * DHEAD + d) * SEQ_N + n] = val;   // transposed
      }
    }
  }
}

// ---------------------------------------------------------------------------
// Kernel 2: flash attention. Block = 128 threads (4 waves) on one (b,h).
// Each wave owns 16 query rows; block covers 64. 32-key chunks double-buffered
// in LDS (K row-major, V transposed); P staged via LDS to become a WMMA
// A-fragment.
// ---------------------------------------------------------------------------
__global__ __launch_bounds__(128) void attn_kernel(
    const _Float16* __restrict__ qbuf, const _Float16* __restrict__ kbuf,
    const _Float16* __restrict__ vbuf, _Float16* __restrict__ obuf) {
  __shared__ _Float16 Ks[2][32][72];     // 32 keys x 64 dh (+8 pad)
  __shared__ _Float16 Vt[2][64][40];     // V^T: 64 dh x 32 keys (+8 pad)
  __shared__ _Float16 Ps[4][16][40];     // per-wave P tile 16x32 (+8 pad)

  const int t = threadIdx.x;
  const int lane = t & 31, wave = t >> 5;
  const int bh = blockIdx.x >> 4;        // b*12 + h
  const int qc = blockIdx.x & 15;        // 64-row query chunk
  const size_t base = (size_t)bh * SEQ_N * DHEAD;
  const int q0 = qc * 64 + wave * 16;

  const v16h qa0 = load_a_frag(qbuf + base, q0, DHEAD, 0, lane);
  const v16h qa1 = load_a_frag(qbuf + base, q0, DHEAD, 32, lane);

  float mrow[8], lrow[8];
  v8f o[4];
  {
    v8f z = {};
#pragma unroll
    for (int i = 0; i < 4; ++i) o[i] = z;
#pragma unroll
    for (int r = 0; r < 8; ++r) { mrow[r] = -1e30f; lrow[r] = 0.0f; }
  }

  auto stageKV = [&](int j, int buf) {
#pragma unroll
    for (int c = 0; c < 2; ++c) {
      const int lin = c * 128 + t;
      const int row = lin >> 3, col = (lin & 7) * 8;
      *(h8*)&Ks[buf][row][col] =
          *(const h8*)(kbuf + base + (size_t)(j * 32 + row) * DHEAD + col);
    }
#pragma unroll
    for (int c = 0; c < 2; ++c) {
      const int lin = c * 128 + t;
      const int d = lin >> 2, col = (lin & 3) * 8;
      *(h8*)&Vt[buf][d][col] =
          *(const h8*)(vbuf + base + (size_t)d * SEQ_N + j * 32 + col);
    }
  };

  stageKV(0, 0);
  for (int j = 0; j < 32; ++j) {         // 32 chunks of 32 keys
    const int cur = j & 1;
    __syncthreads();                     // staging of `cur` complete
    if (j + 1 < 32) stageKV(j + 1, cur ^ 1);

    // S = Q K^T : this wave's 16 rows x 32 keys (two 16x16 tiles).
    v8f s0 = {}, s1 = {};
    {
      const v16h b00 = load_bT_frag(&Ks[cur][0][0], 0, 72, 0, lane);
      const v16h b01 = load_bT_frag(&Ks[cur][0][0], 0, 72, 32, lane);
      const v16h b10 = load_bT_frag(&Ks[cur][0][0], 16, 72, 0, lane);
      const v16h b11 = load_bT_frag(&Ks[cur][0][0], 16, 72, 32, lane);
      s0 = wmma_f16(qa0, b00, s0);
      s0 = wmma_f16(qa1, b01, s0);
      s1 = wmma_f16(qa0, b10, s1);
      s1 = wmma_f16(qa1, b11, s1);
    }

    // Online softmax. C-layout: VGPR r holds row (r + 8*half), col = lane&15.
    _Float16* pw = &Ps[wave][0][0];
    const int cn = lane & 15, half = lane >> 4;
#pragma unroll
    for (int r = 0; r < 8; ++r) {
      float tm = fmaxf(s0[r], s1[r]);
#pragma unroll
      for (int msk = 1; msk <= 8; msk <<= 1)
        tm = fmaxf(tm, __shfl_xor(tm, msk, 32));
      const float mnew = fmaxf(mrow[r], tm);
      const float alpha = __expf(mrow[r] - mnew);
      const float p0 = __expf(s0[r] - mnew);
      const float p1 = __expf(s1[r] - mnew);
      float ps = p0 + p1;
#pragma unroll
      for (int msk = 1; msk <= 8; msk <<= 1) ps += __shfl_xor(ps, msk, 32);
      lrow[r] = lrow[r] * alpha + ps;
      mrow[r] = mnew;
#pragma unroll
      for (int i = 0; i < 4; ++i) o[i][r] *= alpha;
      const int prow = r + 8 * half;
      pw[prow * 40 + cn]      = (_Float16)p0;
      pw[prow * 40 + 16 + cn] = (_Float16)p1;
    }
    __syncthreads();                     // P visible (DS drained)

    // O += P @ V : K-dim = 32 keys; 4 d-tiles from V^T (contiguous reads).
    const v16h pa = load_a_frag(&Ps[wave][0][0], 0, 40, 0, lane);
    v16h vb[4];
#pragma unroll
    for (int dt = 0; dt < 4; ++dt)
      vb[dt] = load_bT_frag(&Vt[cur][0][0], dt * 16, 40, 0, lane);
#pragma unroll
    for (int dt = 0; dt < 4; ++dt) o[dt] = wmma_f16(pa, vb[dt], o[dt]);
  }

  // Normalize, write (B,N,768) f16 for the projection GEMM.
  const int head = bh % NHEADS, b = bh / NHEADS;
#pragma unroll
  for (int dt = 0; dt < 4; ++dt) {
#pragma unroll
    for (int r = 0; r < 8; ++r) {
      const int row = q0 + r + 8 * (lane >> 4);
      const int col = head * DHEAD + dt * 16 + (lane & 15);
      const float inv = 1.0f / lrow[r];
      obuf[(size_t)(b * SEQ_N + row) * DMODEL + col] =
          (_Float16)(o[dt][r] * inv);
    }
  }
}

// ---------------------------------------------------------------------------
// Kernel 3: output projection. Tile 256(M) x 64(N), A already f16.
// Double-buffered LDS like kernel 1.
// ---------------------------------------------------------------------------
__global__ __launch_bounds__(256) void proj_gemm_kernel(
    const _Float16* __restrict__ a, const float* __restrict__ w,
    const float* __restrict__ bias, float* __restrict__ out) {
  __shared__ _Float16 Xs[2][256][40];
  __shared__ _Float16 Wt[2][64][40];     // transposed W tiles

  const int t = threadIdx.x;
  const int lane = t & 31, wave = t >> 5;
  const int m0 = blockIdx.y * 256;
  const int n0 = blockIdx.x * 64;

  v8f acc[2][4];
  {
    v8f z = {};
#pragma unroll
    for (int s = 0; s < 2; ++s)
#pragma unroll
      for (int i = 0; i < 4; ++i) acc[s][i] = z;
  }

  auto stage = [&](int k0, int buf) {
#pragma unroll
    for (int c = 0; c < 4; ++c) {
      const int lin = c * 256 + t;
      const int row = lin >> 2, col = (lin & 3) * 8;
      *(h8*)&Xs[buf][row][col] =
          *(const h8*)(a + (size_t)(m0 + row) * DMODEL + k0 + col);
    }
    const int n = t & 63, kq = (t >> 6) * 8;
    h8 hv;
#pragma unroll
    for (int i = 0; i < 8; ++i)
      hv[i] = (_Float16)w[(size_t)(k0 + kq + i) * DMODEL + n0 + n];
    *(h8*)&Wt[buf][n][kq] = hv;
    if (k0 + 64 < DMODEL)
      __builtin_prefetch(a + (size_t)(m0 + t) * DMODEL + k0 + 64, 0, 1);
  };

  stage(0, 0);
  for (int kk = 0; kk < DMODEL / 32; ++kk) {
    const int cur = kk & 1;
    __syncthreads();
    if (kk + 1 < DMODEL / 32) stage((kk + 1) * 32, cur ^ 1);

    const v16h a0 = load_a_frag(&Xs[cur][0][0], wave * 32, 40, 0, lane);
    const v16h a1 = load_a_frag(&Xs[cur][0][0], wave * 32 + 16, 40, 0, lane);
    v16h bfr[4];
#pragma unroll
    for (int tn = 0; tn < 4; ++tn)
      bfr[tn] = load_bT_frag(&Wt[cur][0][0], tn * 16, 40, 0, lane);
#pragma unroll
    for (int tn = 0; tn < 4; ++tn) {
      acc[0][tn] = wmma_f16(a0, bfr[tn], acc[0][tn]);
      acc[1][tn] = wmma_f16(a1, bfr[tn], acc[1][tn]);
    }
  }

#pragma unroll
  for (int tn = 0; tn < 4; ++tn) {
    const int gcol = n0 + tn * 16 + (lane & 15);
    const float bv = bias[gcol];
#pragma unroll
    for (int s = 0; s < 2; ++s) {
#pragma unroll
      for (int r = 0; r < 8; ++r) {
        const int gm = m0 + wave * 32 + s * 16 + r + 8 * (lane >> 4);
        out[(size_t)gm * DMODEL + gcol] = acc[s][tn][r] + bv;
      }
    }
  }
}

// ---------------------------------------------------------------------------
extern "C" void kernel_launch(void* const* d_in, const int* in_sizes, int n_in,
                              void* d_out, int out_size, void* d_ws,
                              size_t ws_size, hipStream_t stream) {
  (void)in_sizes; (void)n_in; (void)out_size; (void)ws_size;
  const float* x      = (const float*)d_in[0];
  const float* w_qkv  = (const float*)d_in[1];
  const float* b_qkv  = (const float*)d_in[2];
  const float* w_proj = (const float*)d_in[3];
  const float* b_proj = (const float*)d_in[4];
  float* out = (float*)d_out;

  // Workspace: q, k, v (f16) + attn output (B,N,768) f16 = 192 MiB.
  _Float16* qbuf = (_Float16*)d_ws;
  _Float16* kbuf = qbuf + HEADELEMS;
  _Float16* vbuf = kbuf + HEADELEMS;
  _Float16* obuf = vbuf + HEADELEMS;

  dim3 g1(QKV_N / 64, ROWS_M / 256);     // 36 x 128
  qkv_gemm_kernel<<<g1, 256, 0, stream>>>(x, w_qkv, b_qkv, qbuf, kbuf, vbuf);

  attn_kernel<<<BATCH * NHEADS * (SEQ_N / 64), 128, 0, stream>>>(qbuf, kbuf,
                                                                 vbuf, obuf);

  dim3 g3(DMODEL / 64, ROWS_M / 256);    // 12 x 128
  proj_gemm_kernel<<<g3, 256, 0, stream>>>(obuf, w_proj, b_proj, out);
}
